// SeqEncoder_62019327754322
// MI455X (gfx1250) — matrix operand order repaired and checked by reading, not compiled, not run on hardware
//
#include <hip/hip_runtime.h>
#include <stdint.h>

typedef __attribute__((ext_vector_type(16))) __bf16 v16bf;
typedef __attribute__((ext_vector_type(8)))  float  v8f;
typedef __attribute__((ext_vector_type(4)))  int    v4i_t;

#ifndef __has_builtin
#define __has_builtin(x) 0
#endif
#if __has_builtin(__builtin_amdgcn_global_load_async_to_lds_b128)
#define USE_ASYNC_LDS 1
#else
#define USE_ASYNC_LDS 0
#endif

#define B_SZ 256
#define T_SZ 100
#define F_SZ 34
#define E_SZ 1024
#define G_SZ 4096  // 4*E

__device__ __forceinline__ uint16_t f2bf(float f) {
  uint32_t u = __float_as_uint(f);
  uint32_t r = (u + 0x7FFFu + ((u >> 16) & 1u)) >> 16;
  return (uint16_t)r;
}
__device__ __forceinline__ float sigm(float x) { return 1.f / (1.f + __expf(-x)); }

__device__ __forceinline__ void wait_async0() {
#if USE_ASYNC_LDS
#if __has_builtin(__builtin_amdgcn_s_wait_asynccnt)
  __builtin_amdgcn_s_wait_asynccnt(0);
#else
  asm volatile("s_wait_asynccnt 0x0" ::: "memory");
#endif
#endif
}

// ---------------- packing kernels (run once per launch; deterministic) ---------

// Wa[4096][1088] = [ W_hh1 (K=1024) | W_c = W_ih1 @ W_lin (K=34, padded to 64) ]  (bf16)
__global__ __launch_bounds__(256) void pack_wa(const float* __restrict__ W_hh1,
                                               const float* __restrict__ W_ih1,
                                               const float* __restrict__ W_lin,
                                               uint16_t* __restrict__ Wa) {
  size_t gid = (size_t)blockIdx.x * 256 + threadIdx.x;
  if (gid >= (size_t)G_SZ * 1088) return;
  int n = (int)(gid / 1088);
  int k = (int)(gid % 1088);
  float v;
  if (k < 1024) {
    v = W_hh1[(size_t)n * E_SZ + k];
  } else {
    int f = k - 1024;
    if (f < F_SZ) {
      float acc = 0.f;
      const float* wr = W_ih1 + (size_t)n * E_SZ;
      for (int e = 0; e < E_SZ; ++e) acc += wr[e] * W_lin[(size_t)e * F_SZ + f];
      v = acc;
    } else {
      v = 0.f;
    }
  }
  Wa[gid] = f2bf(v);
}

// ba[n] = b_ih1[n] + b_hh1[n] + W_ih1[n,:] . b_lin
__global__ __launch_bounds__(256) void pack_ba(const float* __restrict__ W_ih1,
                                               const float* __restrict__ b_lin,
                                               const float* __restrict__ b_ih1,
                                               const float* __restrict__ b_hh1,
                                               float* __restrict__ ba) {
  int n = blockIdx.x * 256 + threadIdx.x;
  if (n >= G_SZ) return;
  float acc = b_ih1[n] + b_hh1[n];
  const float* wr = W_ih1 + (size_t)n * E_SZ;
  for (int e = 0; e < E_SZ; ++e) acc += wr[e] * b_lin[e];
  ba[n] = acc;
}

// Wb[4096][2048] = [ W_ih2 | W_hh2 ]  (bf16)
__global__ __launch_bounds__(256) void pack_wb(const float* __restrict__ W_ih2,
                                               const float* __restrict__ W_hh2,
                                               uint16_t* __restrict__ Wb) {
  size_t gid = (size_t)blockIdx.x * 256 + threadIdx.x;
  if (gid >= (size_t)G_SZ * 2048) return;
  int n = (int)(gid / 2048);
  int k = (int)(gid % 2048);
  float v = (k < 1024) ? W_ih2[(size_t)n * E_SZ + k]
                       : W_hh2[(size_t)n * E_SZ + (k - 1024)];
  Wb[gid] = f2bf(v);
}

__global__ __launch_bounds__(256) void pack_bb(const float* __restrict__ b_ih2,
                                               const float* __restrict__ b_hh2,
                                               float* __restrict__ bb) {
  int n = blockIdx.x * 256 + threadIdx.x;
  if (n >= G_SZ) return;
  bb[n] = b_ih2[n] + b_hh2[n];
}

// xpad[t][b][64] : first 34 = x[b][t][:], rest zero  (bf16, time-major)
__global__ __launch_bounds__(256) void pack_x(const float* __restrict__ x,
                                              uint16_t* __restrict__ xpad) {
  size_t gid = (size_t)blockIdx.x * 256 + threadIdx.x;
  if (gid >= (size_t)T_SZ * B_SZ * 64) return;
  int fp = (int)(gid & 63);
  size_t tb = gid >> 6;
  int b = (int)(tb % B_SZ);
  int t = (int)(tb / B_SZ);
  float v = (fp < F_SZ) ? x[((size_t)b * T_SZ + t) * F_SZ + fp] : 0.f;
  xpad[gid] = f2bf(v);
}

__global__ __launch_bounds__(256) void init_state(uint16_t* __restrict__ h1,
                                                  uint16_t* __restrict__ h2,
                                                  float* __restrict__ c1,
                                                  float* __restrict__ c2) {
  int i = blockIdx.x * 256 + threadIdx.x;
  if (i < B_SZ * E_SZ) { h1[i] = 0; h2[i] = 0; c1[i] = 0.f; c2[i] = 0.f; }
}

// ---------------- fused LSTM-cell step: GEMM (bf16 WMMA) + gate epilogue -------
// gates[256,4096] = [A0 | A1] @ W^T + bias ; then LSTM update.
// Block = 8 waves: 32 batch rows x 64 hidden cols x all 4 gates.
// Each wave: 2 N-tiles x 2 M-tiles = 4 WMMAs per k-step from 2 B-fragment loads
// (B reuse across M halves the L2 weight traffic per FLOP).
template <int KTOT>
__global__ __launch_bounds__(256) void lstm_step(
    const uint16_t* __restrict__ A0,   // [256,1024] bf16, K segment 0..1023
    const uint16_t* __restrict__ A1,   // [256,lda1] bf16, K segment 1024..KTOT-1
    int lda1,
    const uint16_t* __restrict__ W,    // [4096,KTOT] bf16 row-major
    const float* __restrict__ bias,    // [4096]
    float* __restrict__ c_state,       // [256,1024] in/out (element-exclusive)
    uint16_t* __restrict__ h_out,      // [256,1024] bf16
    float* __restrict__ h_out_f32,     // non-null only on final step
    float* __restrict__ c_out_f32) {   // non-null only on final step
  constexpr int KSTEPS = KTOT / 32;
  __shared__ __attribute__((aligned(16))) uint32_t sA[2][32][16]; // 32 rows x 32 bf16, dbl-buffered
  __shared__ float sG[4][32][64];                                 // gate accumulators

  const int tid  = threadIdx.x;
  const int lane = tid & 31;
  const int wave = tid >> 5;
  const int m0 = blockIdx.x * 32;      // batch-row tile (32 rows)
  const int e0 = blockIdx.y * 64;      // hidden-col tile

  auto stageA = [&](int kt, int buf) {
#if USE_ASYNC_LDS
    // 128 lanes issue one async b128 (8 bf16) each: memory -> LDS, no VGPR data.
    if (tid < 128) {
      int row = tid >> 2;              // 0..31
      int c16 = tid & 3;               // 16-byte chunk
      int gk = kt * 32 + c16 * 8;      // element offset in K
      const uint16_t* src = (gk < 1024)
          ? (A0 + (size_t)(m0 + row) * 1024 + gk)
          : (A1 + (size_t)(m0 + row) * lda1 + (gk - 1024));
      typedef __attribute__((address_space(1))) v4i_t gv4i_t;
      typedef __attribute__((address_space(3))) v4i_t lv4i_t;
      __builtin_amdgcn_global_load_async_to_lds_b128(
          (gv4i_t*)(uintptr_t)src,
          (lv4i_t*)(uint32_t)(uintptr_t)&sA[buf][row][c16 * 4],
          0, 0);
    }
#else
    int row = tid >> 3;                // 0..31
    int c8 = tid & 7;                  // 8-byte chunk (4 bf16)
    int gk = kt * 32 + c8 * 4;
    const uint16_t* src = (gk < 1024)
        ? (A0 + (size_t)(m0 + row) * 1024 + gk)
        : (A1 + (size_t)(m0 + row) * lda1 + (gk - 1024));
    *(uint2*)&sA[buf][row][c8 * 2] = *(const uint2*)src;
#endif
  };

  // wave -> two WMMA output N-tiles: tile t => gate q = t>>2, sub-col s = t&3
  const int t0 = wave * 2, t1 = wave * 2 + 1;
  const int q0 = t0 >> 2, s0 = t0 & 3;
  const int q1 = t1 >> 2, s1 = t1 & 3;
  const size_t n0 = (size_t)(q0 * 1024 + e0 + s0 * 16 + (lane & 15));
  const size_t n1 = (size_t)(q1 * 1024 + e0 + s1 * 16 + (lane & 15));
  const int kbB = (lane >> 4) * 8;     // B-fragment K-octet select (table order)
  const int am  = lane & 15;           // A-fragment row within M-subtile
  const int abA = (lane >> 4) * 4;     // A-fragment K-half select (u32 units)

  v8f acc00 = {};  // M-sub 0, tile t0
  v8f acc01 = {};  // M-sub 0, tile t1
  v8f acc10 = {};  // M-sub 1, tile t0
  v8f acc11 = {};  // M-sub 1, tile t1

  stageA(0, 0);
  wait_async0();
  __syncthreads();

  for (int kt = 0; kt < KSTEPS; ++kt) {
    int buf = kt & 1;
    if (kt + 1 < KSTEPS) stageA(kt + 1, buf ^ 1);

    int gk = kt * 32;
    const uint16_t* wp0 = W + n0 * KTOT + gk + kbB;
    const uint16_t* wp1 = W + n1 * KTOT + gk + kbB;
    __builtin_prefetch(wp0 + 128, 0, 1);   // global_prefetch ahead on weight rows
    __builtin_prefetch(wp1 + 128, 0, 1);

    union { v16bf v; uint32_t u[8]; } af0, af1;
#pragma unroll
    for (int j = 0; j < 8; ++j) {
      int c = ((j >> 2) << 3) + abA + (j & 3);
      af0.u[j] = sA[buf][am][c];
      af1.u[j] = sA[buf][16 + am][c];
    }

    union { v16bf v; uint4 q[2]; } bf0, bf1;
    bf0.q[0] = *(const uint4*)(wp0);
    bf0.q[1] = *(const uint4*)(wp0 + 16);
    bf1.q[0] = *(const uint4*)(wp1);
    bf1.q[1] = *(const uint4*)(wp1 + 16);

    acc00 = __builtin_amdgcn_wmma_f32_16x16x32_bf16(false, af0.v, false, bf0.v,
                                                    (short)0, acc00, false, false);
    acc01 = __builtin_amdgcn_wmma_f32_16x16x32_bf16(false, af0.v, false, bf1.v,
                                                    (short)0, acc01, false, false);
    acc10 = __builtin_amdgcn_wmma_f32_16x16x32_bf16(false, af1.v, false, bf0.v,
                                                    (short)0, acc10, false, false);
    acc11 = __builtin_amdgcn_wmma_f32_16x16x32_bf16(false, af1.v, false, bf1.v,
                                                    (short)0, acc11, false, false);
    wait_async0();
    __syncthreads();
  }

  // C-fragment layout: lane<16 -> M=j, lane>=16 -> M=j+8 ; N = lane&15
#pragma unroll
  for (int j = 0; j < 8; ++j) {
    int m = (lane < 16) ? j : (j + 8);
    sG[q0][m][s0 * 16 + (lane & 15)]      = acc00[j];
    sG[q1][m][s1 * 16 + (lane & 15)]      = acc01[j];
    sG[q0][16 + m][s0 * 16 + (lane & 15)] = acc10[j];
    sG[q1][16 + m][s1 * 16 + (lane & 15)] = acc11[j];
  }
  __syncthreads();

  // fused LSTM gate epilogue: 32x64 = 2048 elements, 8 per thread
#pragma unroll
  for (int r = 0; r < 8; ++r) {
    int idx = tid + r * 256;
    int m = idx >> 6, el = idx & 63;
    int eg = e0 + el;
    size_t off = (size_t)(m0 + m) * 1024 + eg;
    float gi = sG[0][m][el] + bias[eg];
    float gf = sG[1][m][el] + bias[1024 + eg];
    float gg = sG[2][m][el] + bias[2048 + eg];
    float go = sG[3][m][el] + bias[3072 + eg];
    float cn = sigm(gf) * c_state[off] + sigm(gi) * tanhf(gg);
    float hn = sigm(go) * tanhf(cn);
    c_state[off] = cn;
    h_out[off] = f2bf(hn);
    if (h_out_f32) { h_out_f32[off] = hn; c_out_f32[off] = cn; }
  }
}

// ---------------- host orchestration ------------------------------------------
extern "C" void kernel_launch(void* const* d_in, const int* in_sizes, int n_in,
                              void* d_out, int out_size, void* d_ws, size_t ws_size,
                              hipStream_t stream) {
  (void)in_sizes; (void)n_in; (void)out_size; (void)ws_size;
  const float* x     = (const float*)d_in[0];
  const float* W_lin = (const float*)d_in[1];
  const float* b_lin = (const float*)d_in[2];
  const float* W_ih1 = (const float*)d_in[3];
  const float* W_hh1 = (const float*)d_in[4];
  const float* b_ih1 = (const float*)d_in[5];
  const float* b_hh1 = (const float*)d_in[6];
  const float* W_ih2 = (const float*)d_in[7];
  const float* W_hh2 = (const float*)d_in[8];
  const float* b_ih2 = (const float*)d_in[9];
  const float* b_hh2 = (const float*)d_in[10];

  char* ws = (char*)d_ws;
  size_t off = 0;
  auto alloc = [&](size_t bytes) { char* p = ws + off; off += (bytes + 255) & ~(size_t)255; return p; };
  uint16_t* Wa   = (uint16_t*)alloc((size_t)G_SZ * 1088 * 2);
  uint16_t* Wb   = (uint16_t*)alloc((size_t)G_SZ * 2048 * 2);
  uint16_t* xpad = (uint16_t*)alloc((size_t)T_SZ * B_SZ * 64 * 2);
  float*    ba   = (float*)alloc(G_SZ * 4);
  float*    bb   = (float*)alloc(G_SZ * 4);
  uint16_t* h1a  = (uint16_t*)alloc((size_t)B_SZ * E_SZ * 2);
  uint16_t* h1b  = (uint16_t*)alloc((size_t)B_SZ * E_SZ * 2);
  uint16_t* h2a  = (uint16_t*)alloc((size_t)B_SZ * E_SZ * 2);
  uint16_t* h2b  = (uint16_t*)alloc((size_t)B_SZ * E_SZ * 2);
  float*    c1   = (float*)alloc((size_t)B_SZ * E_SZ * 4);
  float*    c2   = (float*)alloc((size_t)B_SZ * E_SZ * 4);

  pack_wa<<<(unsigned)(((size_t)G_SZ * 1088 + 255) / 256), 256, 0, stream>>>(W_hh1, W_ih1, W_lin, Wa);
  pack_wb<<<(unsigned)(((size_t)G_SZ * 2048 + 255) / 256), 256, 0, stream>>>(W_ih2, W_hh2, Wb);
  pack_ba<<<(G_SZ + 255) / 256, 256, 0, stream>>>(W_ih1, b_lin, b_ih1, b_hh1, ba);
  pack_bb<<<(G_SZ + 255) / 256, 256, 0, stream>>>(b_ih2, b_hh2, bb);
  pack_x<<<(unsigned)(((size_t)T_SZ * B_SZ * 64 + 255) / 256), 256, 0, stream>>>(x, xpad);
  init_state<<<(B_SZ * E_SZ + 255) / 256, 256, 0, stream>>>(h1a, h2a, c1, c2);

  dim3 grid(8, 16);  // 8 batch-row tiles (32 rows) x 16 hidden-col tiles
  uint16_t *h1c = h1a, *h1n = h1b, *h2c = h2a, *h2n = h2b;
  for (int t = 0; t < T_SZ; ++t) {
    // layer 1: gates = [h1 | x_t] @ [W_hh1 | W_c]^T + ba
    lstm_step<1088><<<grid, 256, 0, stream>>>(h1c, xpad + (size_t)t * B_SZ * 64, 64,
                                              Wa, ba, c1, h1n, nullptr, nullptr);
    // layer 2: gates = [h1_new | h2] @ [W_ih2 | W_hh2]^T + bb
    bool last = (t == T_SZ - 1);
    lstm_step<2048><<<grid, 256, 0, stream>>>(h1n, h2c, 1024, Wb, bb, c2, h2n,
                                              last ? (float*)d_out : nullptr,
                                              last ? (float*)d_out + (size_t)B_SZ * E_SZ : nullptr);
    uint16_t* tmp;
    tmp = h1c; h1c = h1n; h1n = tmp;
    tmp = h2c; h2c = h2n; h2n = tmp;
  }
}
